// CausalAttentiveStatisticsPooling_25116968747080
// MI455X (gfx1250) — matrix operand order repaired and checked by reading, not compiled, not run on hardware
//
#include <hip/hip_runtime.h>
#include <math.h>

#define B_   8
#define C_   1536
#define T_   4096
#define A_   128
#define K3C  4608
#define EPSF 1e-12f

typedef __attribute__((ext_vector_type(16))) __bf16 bf16x16;
typedef __attribute__((ext_vector_type(8)))  float  f32x8;
struct U128x2 { uint4 lo, hi; };

// ---------- helpers ----------
__device__ __forceinline__ unsigned short f2bf(float f) {
  unsigned u = __builtin_bit_cast(unsigned, f);
  u += 0x7FFFu + ((u >> 16) & 1u);            // round-to-nearest-even
  return (unsigned short)(u >> 16);
}

// Workgroup split-barrier with ONLY the LDS-visibility wait (s_wait_dscnt).
// Unlike __syncthreads(), this does not fence global memory, so in-flight
// global loads/stores keep flying across the barrier (true SW pipelining).
__device__ __forceinline__ void wg_barrier() {
  asm volatile("s_wait_dscnt 0x0\n\t"
               "s_barrier_signal -1\n\t"
               "s_barrier_wait -1" ::: "memory");
}

__device__ __forceinline__ float wave_incl_scan(float v, int lane) {
#pragma unroll
  for (int d = 1; d < 32; d <<= 1) {
    float n = __shfl_up(v, d, 32);
    if (lane >= d) v += n;
  }
  return v;
}
__device__ __forceinline__ float wave_bcast31(float v) { return __shfl(v, 31, 32); }
__device__ __forceinline__ float wave_sum(float v) {
#pragma unroll
  for (int m = 16; m >= 1; m >>= 1) v += __shfl_xor(v, m, 32);
  return v;
}

// ---------- k0: W1 fp32 -> bf16 ----------
__global__ __launch_bounds__(256) void k0_prep(const float* __restrict__ W1,
                                               unsigned short* __restrict__ W1b) {
  int i = blockIdx.x * 256 + threadIdx.x;
  if (i < A_ * K3C) W1b[i] = f2bf(W1[i]);
}

// ---------- k1: causal mean/std scan + bf16 transpose into attnT[b][t][k] ----------
__global__ __launch_bounds__(256) void k1_causal(const float* __restrict__ x,
                                                 const int* __restrict__ lengths,
                                                 unsigned short* __restrict__ attnT) {
  __shared__ unsigned short tile[3 * 32 * 40];   // [part][t(32)][c(32), stride 40]
  const int tid  = threadIdx.x;
  const int lane = tid & 31, w = tid >> 5;
  const int b  = blockIdx.x / (C_ / 32);
  const int c0 = (blockIdx.x % (C_ / 32)) * 32;
  const int len = lengths[b];
  const float flen = (float)len;
  float S1[4] = {0.f, 0.f, 0.f, 0.f};
  float S2[4] = {0.f, 0.f, 0.f, 0.f};
  const float* xb = x + (size_t)b * C_ * T_;
  unsigned short* dstBase = attnT + (size_t)b * T_ * K3C;

  for (int t0 = 0; t0 < T_; t0 += 32) {
    const int t = t0 + lane;
    const float cnt = fminf((float)(t + 1), flen);   // len >= T/2 >= 1
#pragma unroll
    for (int r = 0; r < 4; ++r) {
      const int c = c0 + w * 4 + r;
      float v  = xb[(size_t)c * T_ + t];
      float vm = (t < len) ? v : 0.f;
      float s1 = wave_incl_scan(vm, lane) + S1[r];
      float s2 = wave_incl_scan(vm * vm, lane) + S2[r];
      S1[r] = wave_bcast31(s1);
      S2[r] = wave_bcast31(s2);
      float mean = s1 / cnt;
      float var  = s2 / cnt - mean * mean;
      float sd   = sqrtf(fmaxf(var, EPSF));
      const int cw = w * 4 + r;
      tile[(0 * 32 + lane) * 40 + cw] = f2bf(v);
      tile[(1 * 32 + lane) * 40 + cw] = f2bf(mean);
      tile[(2 * 32 + lane) * 40 + cw] = f2bf(sd);
    }
    wg_barrier();
    for (int j = tid; j < 384; j += 256) {          // 3 parts * 32 t * 4 chunks(16B)
      int part = j / 128, rem = j % 128;
      int tt = rem >> 2, chunk = rem & 3;
      uint4 val = *(const uint4*)&tile[(part * 32 + tt) * 40 + chunk * 8];
      *(uint4*)&dstBase[(size_t)(t0 + tt) * K3C + part * C_ + c0 + chunk * 8] = val;
    }
    wg_barrier();   // ds_loads for out-copy complete (dscnt); stores drain async
  }
}

// ---------- k2: bf16 WMMA GEMM (64t x 128a per block) + tanh + W2 dot -> logits ----------
// Register-pipelined: next K-chunk (A tile + per-wave B frags) loads overlap WMMAs
// and now also the barriers (wg_barrier doesn't drain loadcnt).
__global__ __launch_bounds__(256) void k2_gemm(const unsigned short* __restrict__ attnT,
                                               const unsigned short* __restrict__ W1b,
                                               const float* __restrict__ b1,
                                               const float* __restrict__ W2,
                                               float* __restrict__ logits) {
  __shared__ unsigned short ldsA[64 * 72];    // 64 t-rows x 64 k (row stride 144B)
  __shared__ float lpart[8][64];
  const int tid  = threadIdx.x;
  const int lane = tid & 31, w = tid >> 5;
  const int b     = blockIdx.x / (T_ / 64);
  const int tbase = (blockIdx.x % (T_ / 64)) * 64;
  const unsigned short* Abase = attnT + ((size_t)b * T_ + tbase) * K3C;

  f32x8 acc[4] = {};
  const int hrow = lane & 15;   // A: t-row within 16 / B: a-column
  const int hhi  = lane >> 4;   // lane half selects K sub-range

  // A staging: thread -> row tid>>2, 32 contiguous bytes at (tid&3)*32
  const int arow = tid >> 2;
  const int achk = (tid & 3) * 2;                       // chunk pair (16B each)
  const unsigned short* Ag = Abase + (size_t)arow * K3C + achk * 8;
  unsigned short*       Al = &ldsA[arow * 72 + achk * 8];
  // B fragments: lane n -> row w*16 + n%16; K bytes offset by lane half
  const unsigned short* Bg = W1b + (size_t)(w * 16 + hrow) * K3C + hhi * 16;

  // preload K-chunk 0 (A first so the ds_store wait can retire earliest)
  uint4 ar0 = *(const uint4*)(Ag + 0);
  uint4 ar1 = *(const uint4*)(Ag + 8);
  uint4 bn0 = *(const uint4*)(Bg + 0);
  uint4 bn1 = *(const uint4*)(Bg + 8);
  uint4 bn2 = *(const uint4*)(Bg + 32);
  uint4 bn3 = *(const uint4*)(Bg + 40);

  for (int k0 = 0; k0 < K3C; k0 += 64) {
    *(uint4*)(Al + 0) = ar0;
    *(uint4*)(Al + 8) = ar1;
    uint4 bc0 = bn0, bc1 = bn1, bc2 = bn2, bc3 = bn3;
    wg_barrier();                                      // ldsA filled

    // prefetch next K-chunk (overlaps the 8 WMMAs and the closing barrier)
    const int kn = (k0 + 64 < K3C) ? (k0 + 64) : k0;   // last iter: harmless reload
    ar0 = *(const uint4*)(Ag + kn);
    ar1 = *(const uint4*)(Ag + kn + 8);
    bn0 = *(const uint4*)(Bg + kn);
    bn1 = *(const uint4*)(Bg + kn + 8);
    bn2 = *(const uint4*)(Bg + kn + 32);
    bn3 = *(const uint4*)(Bg + kn + 40);

#pragma unroll
    for (int sub = 0; sub < 2; ++sub) {
      U128x2 bt;
      bt.lo = sub ? bc2 : bc0;
      bt.hi = sub ? bc3 : bc1;
      bf16x16 bfrag = __builtin_bit_cast(bf16x16, bt);
#pragma unroll
      for (int s = 0; s < 4; ++s) {
        // A frag: v0-7 = K0-15 / K16-31 (lane half), contiguous like B
        U128x2 at;
        const unsigned short* p = &ldsA[(s * 16 + hrow) * 72 + sub * 32 + hhi * 16];
        at.lo = *(const uint4*)p;
        at.hi = *(const uint4*)(p + 8);
        bf16x16 afrag = __builtin_bit_cast(bf16x16, at);
        acc[s] = __builtin_amdgcn_wmma_f32_16x16x32_bf16(
            false, afrag, false, bfrag, (short)0, acc[s], false, false);
      }
    }
    wg_barrier();                                      // done reading ldsA
  }

  // epilogue: logits[t] = sum_a W2[a]*tanh(D[t][a] + b1[a])  (b2 cancels in softmax)
  const int a = w * 16 + hrow;
  const float b1a = b1[a];
  const float w2a = W2[a];
#pragma unroll
  for (int s = 0; s < 4; ++s) {
#pragma unroll
    for (int j = 0; j < 8; ++j) {
      float hv = tanhf(acc[s][j] + b1a) * w2a;
#pragma unroll
      for (int m = 1; m <= 8; m <<= 1) hv += __shfl_xor(hv, m, 32);  // reduce 16 a's
      if (hrow == 0) lpart[w][s * 16 + j + hhi * 8] = hv;
    }
  }
  wg_barrier();
  if (tid < 64) {                     // fixed-order cross-wave sum -> deterministic
    float sum = 0.f;
#pragma unroll
    for (int ww = 0; ww < 8; ++ww) sum += lpart[ww][tid];
    logits[(size_t)b * T_ + tbase + tid] = sum;
  }
}

// ---------- k3: per-b softmax weights e and cumulative Z ----------
__global__ __launch_bounds__(256) void k3_softmax(const float* __restrict__ logits,
                                                  float* __restrict__ e,
                                                  float* __restrict__ Z) {
  const int lane = threadIdx.x & 31, b = threadIdx.x >> 5;   // 8 waves = 8 batches
  const float* lg = logits + (size_t)b * T_;
  float mx = -3.402823466e38f;
  for (int t0 = 0; t0 < T_; t0 += 32) mx = fmaxf(mx, lg[t0 + lane]);
#pragma unroll
  for (int m = 16; m >= 1; m >>= 1) mx = fmaxf(mx, __shfl_xor(mx, m, 32));
  float carry = 0.f;
  for (int t0 = 0; t0 < T_; t0 += 32) {
    float ev = expf(lg[t0 + lane] - mx);
    float s  = wave_incl_scan(ev, lane) + carry;
    e[(size_t)b * T_ + t0 + lane] = ev;
    Z[(size_t)b * T_ + t0 + lane] = s;
    carry = wave_bcast31(s);
  }
}

// ---------- k4: weighted running stats scan + time-average ----------
__global__ __launch_bounds__(256) void k4_final(const float* __restrict__ x,
                                                const float* __restrict__ e,
                                                const float* __restrict__ Z,
                                                float* __restrict__ out) {
  const int lane = threadIdx.x & 31, w = threadIdx.x >> 5;
  const int g = blockIdx.x * 8 + w;
  const int b = g / C_, c = g % C_;
  const float* xr = x + ((size_t)b * C_ + c) * T_;
  const float* er = e + (size_t)b * T_;
  const float* zr = Z + (size_t)b * T_;
  float S1 = 0.f, S2 = 0.f, accM = 0.f, accS = 0.f;
  for (int t0 = 0; t0 < T_; t0 += 32) {
    float xv = xr[t0 + lane];
    float ev = er[t0 + lane];
    float zv = zr[t0 + lane];
    float s1 = wave_incl_scan(ev * xv, lane) + S1;
    S1 = wave_bcast31(s1);
    float wm = s1 / zv;                       // inclusive weighted mean at t
    accM += wm;
    float d = xv - wm;
    float s2 = wave_incl_scan(ev * d * d, lane) + S2;
    S2 = wave_bcast31(s2);
    accS += sqrtf(fmaxf(s2 / zv, EPSF));
  }
  accM = wave_sum(accM);
  accS = wave_sum(accS);
  if (lane == 0) {
    const float fw = 1.0f / ((float)T_ + EPSF);
    out[(size_t)b * (2 * C_) + c]      = accM * fw;
    out[(size_t)b * (2 * C_) + C_ + c] = accS * fw;
  }
}

// ---------- launcher ----------
extern "C" void kernel_launch(void* const* d_in, const int* in_sizes, int n_in,
                              void* d_out, int out_size, void* d_ws, size_t ws_size,
                              hipStream_t stream) {
  (void)in_sizes; (void)n_in; (void)out_size; (void)ws_size;
  const float* x       = (const float*)d_in[0];
  const int*   lengths = (const int*)d_in[1];
  const float* W1      = (const float*)d_in[2];
  const float* b1      = (const float*)d_in[3];
  const float* W2      = (const float*)d_in[4];
  // d_in[5] (b2) is shift-invariant under the softmax -> unused.

  const size_t SZ_ATTN = (size_t)B_ * T_ * K3C * sizeof(unsigned short); // 302.0 MB
  const size_t SZ_W1B  = (size_t)A_ * K3C * sizeof(unsigned short);      // 1.18 MB
  const size_t SZ_LOG  = (size_t)B_ * T_ * sizeof(float);                // 128 KB

  char* ws = (char*)d_ws;
  unsigned short* attnT  = (unsigned short*)(ws);
  unsigned short* W1b    = (unsigned short*)(ws + SZ_ATTN);
  float*          logits = (float*)(ws + SZ_ATTN + SZ_W1B);
  float*          eArr   = (float*)(ws + SZ_ATTN + SZ_W1B + SZ_LOG);
  float*          zArr   = (float*)(ws + SZ_ATTN + SZ_W1B + 2 * SZ_LOG);
  float*          outf   = (float*)d_out;

  k0_prep   <<<(A_ * K3C + 255) / 256, 256, 0, stream>>>(W1, W1b);
  k1_causal <<<B_ * (C_ / 32),         256, 0, stream>>>(x, lengths, attnT);
  k2_gemm   <<<B_ * (T_ / 64),         256, 0, stream>>>(attnT, W1b, b1, W2, logits);
  k3_softmax<<<1,                      256, 0, stream>>>(logits, eArr, zArr);
  k4_final  <<<(B_ * C_) / 8,          256, 0, stream>>>(x, eArr, zArr, outf);
}
